// RGCN_51479478010373
// MI455X (gfx1250) — compile-verified, matrix operand order checked by reading, not run on hardware
//
#include <hip/hip_runtime.h>
#include <math.h>

#define NNODE 50000
#define NREL  8
#define DIN   96
#define DHID  16
#define DOUT  32
#define NTILE (NNODE / 16)   // 3125 16-node WMMA tiles

typedef __attribute__((ext_vector_type(16))) _Float16 v16h;
typedef __attribute__((ext_vector_type(8)))  float    v8f;

// K-offset within a 32-wide K chunk for 16-bit A/B fragments (ISA 7.12.2):
// lanes 0-15: VGPR v in 0..3 -> K = 2v,2v+1 ; v in 4..7 -> K = 16+2(v-4),...
// lanes 16-31: same pattern shifted by +8.
__device__ __forceinline__ int kbase_of(int v, int hi) {
  int b = (v < 4) ? (2 * v) : (16 + 2 * (v - 4));
  return b + (hi ? 8 : 0);
}

__device__ __forceinline__ void atomAddF(float* p, float v) {
#if defined(__has_builtin)
#if __has_builtin(__hip_atomic_fetch_add)
  __hip_atomic_fetch_add(p, v, __ATOMIC_RELAXED, __HIP_MEMORY_SCOPE_AGENT);
  return;
#endif
#endif
  atomicAdd(p, v);
}

// ---------------------------------------------------------------- zero scratch
__global__ void k_zero(float4* __restrict__ p, long n4) {
  long i = blockIdx.x * (long)blockDim.x + threadIdx.x;
  long stride = (long)gridDim.x * blockDim.x;
  float4 z = {0.f, 0.f, 0.f, 0.f};
  for (; i < n4; i += stride) p[i] = z;
}

// ------------------------------------------- layer 1: transform-then-aggregate
// 8 threads per edge; thread jj handles output cols jj and jj+8.
// W1_rel (48KB) staged in LDS once per persistent block.
__global__ void k_edge1(const int* __restrict__ src, const int* __restrict__ dst,
                        const int* __restrict__ et, const float* __restrict__ x,
                        const float* __restrict__ W1rel,
                        float* __restrict__ acc1, float* __restrict__ cnt, int E) {
  __shared__ float Ws[NREL * DIN * DHID];  // 12288 floats = 48KB
  for (int i = threadIdx.x; i < NREL * DIN * DHID; i += blockDim.x)
    Ws[i] = W1rel[i];
  __syncthreads();

  int jj = threadIdx.x & 7;
  int elocal = threadIdx.x >> 3;  // 0..31 edges per 256-thread block
  for (int base = blockIdx.x * 32; base < E; base += gridDim.x * 32) {
    int e = base + elocal;
    if (e < E) {
      int s = src[e], d = dst[e], r = et[e];
      const float4* xr = (const float4*)(x + (size_t)s * DIN);  // 384B-aligned row
      const float* w = Ws + r * (DIN * DHID);
      float a0 = 0.f, a1 = 0.f;
#pragma unroll
      for (int i4 = 0; i4 < DIN / 4; ++i4) {
        float4 xv = xr[i4];                 // broadcast across the 8 edge-threads
        const float* wr = w + i4 * 4 * DHID;
        a0 += xv.x * wr[jj]      + xv.y * wr[DHID + jj] +
              xv.z * wr[2*DHID + jj] + xv.w * wr[3*DHID + jj];
        a1 += xv.x * wr[8 + jj]      + xv.y * wr[DHID + 8 + jj] +
              xv.z * wr[2*DHID + 8 + jj] + xv.w * wr[3*DHID + 8 + jj];
      }
      float* dp = acc1 + ((size_t)d * NREL + r) * DHID;
      atomAddF(dp + jj, a0);
      atomAddF(dp + 8 + jj, a1);
      if (jj == 0) atomAddF(&cnt[d * NREL + r], 1.0f);
    }
  }
}

// --------------------------- layer 1 finalize: h = relu(x@W1_root + means + b1)
// One wave32 per 16-node tile; x@W1_root via 3 chained v_wmma_f32_16x16x32_f16.
__global__ void k_finalize1(const float* __restrict__ x,
                            const float* __restrict__ Wroot,
                            const float* __restrict__ bias,
                            const float* __restrict__ acc1,
                            const float* __restrict__ cnt,
                            float* __restrict__ h) {
  int wave = blockIdx.x * (blockDim.x >> 5) + (threadIdx.x >> 5);
  if (wave >= NTILE) return;
  int lane = threadIdx.x & 31;
  int hi = lane >> 4;
  int lr = lane & 15;
  int m0 = wave * 16;
  int mA = m0 + lr;

  v8f c = {};
#pragma unroll
  for (int kc = 0; kc < 3; ++kc) {        // K = 96 = 3 x 32
    v16h a, b;
#pragma unroll
    for (int v = 0; v < 8; ++v) {
      int k = kc * 32 + kbase_of(v, hi);
      float2 xa = *(const float2*)(x + (size_t)mA * DIN + k);
      a[2 * v]     = (_Float16)xa.x;
      a[2 * v + 1] = (_Float16)xa.y;
      b[2 * v]     = (_Float16)Wroot[k * DHID + lr];
      b[2 * v + 1] = (_Float16)Wroot[(k + 1) * DHID + lr];
    }
    c = __builtin_amdgcn_wmma_f32_16x16x32_f16(false, a, false, b,
                                               (short)0, c, false, false);
  }

#pragma unroll
  for (int v = 0; v < 8; ++v) {           // D element: row v+8*hi, col lr
    int m = m0 + v + hi * 8;
    float s = c[v] + bias[lr];
#pragma unroll
    for (int r = 0; r < NREL; ++r) {
      int seg = m * NREL + r;
      float cn = cnt[seg];
      float inv = (cn > 0.f) ? (1.f / cn) : 0.f;  // == /max(cnt,1): cnt==0 -> acc==0
      s += acc1[(size_t)seg * DHID + lr] * inv;
    }
    h[(size_t)m * DHID + lr] = fmaxf(s, 0.f);
  }
}

// ------------------------------------------- layer 2: aggregate-then-transform
__global__ void k_edge2(const int* __restrict__ src, const int* __restrict__ dst,
                        const int* __restrict__ et, const float* __restrict__ hbuf,
                        float* __restrict__ agg2, int E) {
  long total = (long)E * DHID;
  long idx = blockIdx.x * (long)blockDim.x + threadIdx.x;
  long stride = (long)gridDim.x * blockDim.x;
  for (; idx < total; idx += stride) {
    int e = (int)(idx >> 4);
    int j = (int)(idx & 15);
    float v = hbuf[(size_t)src[e] * DHID + j];
    atomAddF(agg2 + ((size_t)dst[e] * NREL + et[e]) * DHID + j, v);
  }
}

// ---- layer 2 finalize: raw = [mean2(n,r,:)||h(n,:)] @ [W2rel-stack||W2root] + b2
// K = 8*16 + 16 = 144, padded to 160 = 5 chunks of 32; N = 32 = two 16-wide halves.
__global__ void k_finalize2(const float* __restrict__ agg2,
                            const float* __restrict__ cnt,
                            const float* __restrict__ hbuf,
                            const float* __restrict__ W2rel,
                            const float* __restrict__ W2root,
                            const float* __restrict__ bias,
                            float* __restrict__ raw) {
  int wave = blockIdx.x * (blockDim.x >> 5) + (threadIdx.x >> 5);
  if (wave >= NTILE) return;
  int lane = threadIdx.x & 31;
  int hi = lane >> 4;
  int lr = lane & 15;
  int m0 = wave * 16;
  int mA = m0 + lr;

  float inv[NREL];
#pragma unroll
  for (int r = 0; r < NREL; ++r) {
    float cn = cnt[mA * NREL + r];
    inv[r] = (cn > 0.f) ? (1.f / cn) : 0.f;
  }

  v8f c0 = {}, c1 = {};
#pragma unroll
  for (int kc = 0; kc < 5; ++kc) {
    v16h a, b0, b1;
#pragma unroll
    for (int v = 0; v < 8; ++v) {
      int k = kbase_of(v, hi);
      int gk = kc * 32 + k;  // compile-time after unroll; branches fold away
      float e0, e1;
      if (gk < 128) {                       // mean over relation r = gk/16
        int r = gk >> 4;
        int i = gk & 15;
        float2 t = *(const float2*)(agg2 + ((size_t)mA * NREL + r) * DHID + i);
        e0 = t.x * inv[r];
        e1 = t.y * inv[r];
      } else if (gk < 144) {                // root part: h row
        float2 t = *(const float2*)(hbuf + (size_t)mA * DHID + (gk - 128));
        e0 = t.x; e1 = t.y;
      } else {                              // zero padding to K=160
        e0 = 0.f; e1 = 0.f;
      }
      a[2 * v]     = (_Float16)e0;
      a[2 * v + 1] = (_Float16)e1;

      float w00, w01, w10, w11;
      if (gk < 128) {
        w00 = W2rel[gk * DOUT + lr];        w10 = W2rel[gk * DOUT + 16 + lr];
        w01 = W2rel[(gk + 1) * DOUT + lr];  w11 = W2rel[(gk + 1) * DOUT + 16 + lr];
      } else if (gk < 144) {
        w00 = W2root[(gk - 128) * DOUT + lr];       w10 = W2root[(gk - 128) * DOUT + 16 + lr];
        w01 = W2root[(gk - 127) * DOUT + lr];       w11 = W2root[(gk - 127) * DOUT + 16 + lr];
      } else {
        w00 = w01 = w10 = w11 = 0.f;
      }
      b0[2 * v] = (_Float16)w00; b0[2 * v + 1] = (_Float16)w01;
      b1[2 * v] = (_Float16)w10; b1[2 * v + 1] = (_Float16)w11;
    }
    c0 = __builtin_amdgcn_wmma_f32_16x16x32_f16(false, a, false, b0,
                                                (short)0, c0, false, false);
    c1 = __builtin_amdgcn_wmma_f32_16x16x32_f16(false, a, false, b1,
                                                (short)0, c1, false, false);
  }

#pragma unroll
  for (int v = 0; v < 8; ++v) {
    int m = m0 + v + hi * 8;
    raw[(size_t)m * DOUT + lr]      = c0[v] + bias[lr];
    raw[(size_t)m * DOUT + 16 + lr] = c1[v] + bias[16 + lr];
  }
}

// ---------------------------------------- row log_softmax: one wave32 per node
__global__ void k_logsoftmax(const float* __restrict__ raw, float* __restrict__ out) {
  int node = blockIdx.x * (blockDim.x >> 5) + (threadIdx.x >> 5);
  if (node >= NNODE) return;
  int lane = threadIdx.x & 31;  // 32 lanes == 32 columns
  float v = raw[(size_t)node * DOUT + lane];
  float m = v;
#pragma unroll
  for (int o = 16; o > 0; o >>= 1) m = fmaxf(m, __shfl_xor(m, o));
  float s = __expf(v - m);
#pragma unroll
  for (int o = 16; o > 0; o >>= 1) s += __shfl_xor(s, o);
  out[(size_t)node * DOUT + lane] = v - m - __logf(s);
}

// -----------------------------------------------------------------------------
extern "C" void kernel_launch(void* const* d_in, const int* in_sizes, int n_in,
                              void* d_out, int out_size, void* d_ws, size_t ws_size,
                              hipStream_t stream) {
  const float* x      = (const float*)d_in[0];
  const int*   eidx   = (const int*)d_in[1];   // [2, E] row-major
  const int*   etype  = (const int*)d_in[2];
  const float* W1rel  = (const float*)d_in[3];
  const float* W1root = (const float*)d_in[4];
  const float* b1     = (const float*)d_in[5];
  const float* W2rel  = (const float*)d_in[6];
  const float* W2root = (const float*)d_in[7];
  const float* b2     = (const float*)d_in[8];
  int E = in_sizes[2];
  const int* src = eidx;
  const int* dst = eidx + E;

  // workspace layout (floats): acc1 | cnt | agg2 | h | raw   (~62.4 MB total)
  float* ws   = (float*)d_ws;
  float* acc1 = ws;                                       // N*R*16 = 6,400,000
  float* cnt  = acc1 + (size_t)NNODE * NREL * DHID;       //   N*R =   400,000
  float* agg2 = cnt  + (size_t)NNODE * NREL;              // N*R*16 = 6,400,000
  float* hbuf = agg2 + (size_t)NNODE * NREL * DHID;       //  N*16 =   800,000
  float* raw  = hbuf + (size_t)NNODE * DHID;              //  N*32 = 1,600,000

  long nzero = (long)NNODE * NREL * DHID * 2 + (long)NNODE * NREL;  // 13,200,000
  k_zero<<<4096, 256, 0, stream>>>((float4*)ws, nzero / 4);

  k_edge1<<<1536, 256, 0, stream>>>(src, dst, etype, x, W1rel, acc1, cnt, E);

  int fblocks = (NTILE + 3) / 4;  // 4 waves (tiles) per 128-thread block
  k_finalize1<<<fblocks, 128, 0, stream>>>(x, W1root, b1, acc1, cnt, hbuf);

  k_edge2<<<4096, 256, 0, stream>>>(src, dst, etype, hbuf, agg2, E);

  k_finalize2<<<fblocks, 128, 0, stream>>>(agg2, cnt, hbuf, W2rel, W2root, b2, raw);

  k_logsoftmax<<<(NNODE + 7) / 8, 256, 0, stream>>>(raw, (float*)d_out);
}
